// TransformerDecoderLayer_63307817943775
// MI455X (gfx1250) — matrix operand order repaired and checked by reading, not compiled, not run on hardware
//
#include <hip/hip_runtime.h>
#include <hip/hip_bf16.h>

typedef __attribute__((ext_vector_type(16))) _Float16 v16h;
typedef __attribute__((ext_vector_type(8)))  _Float16 v8h;
typedef __attribute__((ext_vector_type(8)))  float    v8f;
typedef __attribute__((ext_vector_type(4)))  int      v4i;

#define D_MODEL 1024
#define NHEAD   16
#define DHEAD   64
#define DFF_N   4096
#define BATCH   4
#define SEQ     1024
#define MROWS   (BATCH*SEQ)
#define EPS_    1e-5f

#if __has_builtin(__builtin_amdgcn_global_load_async_to_lds_b128) && \
    __has_builtin(__builtin_amdgcn_s_wait_asynccnt)
#define USE_ASYNC 1
#else
#define USE_ASYNC 0
#endif

__device__ __forceinline__ v8f wmma32(v16h a, v16h b, v8f c) {
  // D = A(16x32 f16) * B(32x16 f16) + C(16x16 f32)
  return __builtin_amdgcn_wmma_f32_16x16x32_f16(false, a, false, b, (short)0, c,
                                                false, false);
}

#if USE_ASYNC
__device__ __forceinline__ void async_b128(const _Float16* g, _Float16* l) {
  __builtin_amdgcn_global_load_async_to_lds_b128(
      (__attribute__((address_space(1))) v4i*)g,
      (__attribute__((address_space(3))) v4i*)l, 0, 0);
}
#endif

// ---- wave32 WMMA fragment gathers: all two b128 LDS loads ------------------
// A fragment: 16(M) x 32(K) tile, row-major, pitch in halfs.
__device__ __forceinline__ v16h frag_a(const _Float16* t, int pitch) {
  int lane = threadIdx.x & 31;
  int hl = lane >> 4, m = lane & 15;
  v8h lo = *(const v8h*)(t + m * pitch + 8 * hl);
  v8h hi = *(const v8h*)(t + m * pitch + 16 + 8 * hl);
  return __builtin_shufflevector(lo, hi, 0, 1, 2, 3, 4, 5, 6, 7,
                                 8, 9, 10, 11, 12, 13, 14, 15);
}
// B fragment from an N-major tile t[n][k] (pitch in halfs): B[k][n] = t[n*pitch+k]
__device__ __forceinline__ v16h frag_b_t(const _Float16* t, int pitch) {
  int lane = threadIdx.x & 31;
  int hl = lane >> 4, n = lane & 15;
  v8h lo = *(const v8h*)(t + n * pitch + 8 * hl);
  v8h hi = *(const v8h*)(t + n * pitch + 16 + 8 * hl);
  return __builtin_shufflevector(lo, hi, 0, 1, 2, 3, 4, 5, 6, 7,
                                 8, 9, 10, 11, 12, 13, 14, 15);
}

// ---- elementwise f32 -> f16 ------------------------------------------------
__global__ void __launch_bounds__(256)
f32_to_f16_k(const float* __restrict__ s, _Float16* __restrict__ d, long n) {
  long i = (long)blockIdx.x * 256 + threadIdx.x;
  if (i < n) d[i] = (_Float16)s[i];
}

// ---- f32 [K,N] -> f16 transposed [N,K] (weights) ---------------------------
__global__ void __launch_bounds__(256)
f32_to_f16_tr(const float* __restrict__ src, _Float16* __restrict__ dst,
              int K, int N) {
  __shared__ _Float16 tile[32][33];
  int k0 = blockIdx.y * 32, n0 = blockIdx.x * 32;
  int tx = threadIdx.x & 31, ty = threadIdx.x >> 5;
#pragma unroll
  for (int i = 0; i < 4; ++i)
    tile[ty + i * 8][tx] = (_Float16)src[(long)(k0 + ty + i * 8) * N + n0 + tx];
  __syncthreads();
#pragma unroll
  for (int i = 0; i < 4; ++i)
    dst[(long)(n0 + ty + i * 8) * K + k0 + tx] = tile[tx][ty + i * 8];
}

// ---- V head transpose: [B*S, H*64] f16 -> [B,H,64,S] f16 -------------------
__global__ void __launch_bounds__(256)
vhead_tr(const _Float16* __restrict__ src, _Float16* __restrict__ dst) {
  __shared__ _Float16 tile[32][33];
  int b = blockIdx.z;
  int s0 = blockIdx.x * 32, c0 = blockIdx.y * 32;
  int tx = threadIdx.x & 31, ty = threadIdx.x >> 5;
#pragma unroll
  for (int i = 0; i < 4; ++i)
    tile[ty + i * 8][tx] =
        src[((long)(b * SEQ + s0 + ty + i * 8)) * D_MODEL + c0 + tx];
  __syncthreads();
#pragma unroll
  for (int i = 0; i < 4; ++i) {
    int c = c0 + ty + i * 8;
    dst[((long)(b * NHEAD + (c >> 6)) * DHEAD + (c & 63)) * SEQ + s0 + tx] =
        tile[tx][ty + i * 8];
  }
}

// ---- WMMA GEMM: out = A[M,K] @ W + bias; W given TRANSPOSED [N,K] ----------
// 256 threads (8 waves); block tile 128x128, K step 32; double-buffered LDS;
// software-pipelined (async-to-LDS when available, else register staging).
#define AP 40   // LDS pitch for 32-wide half tiles (bank spread)
__global__ void __launch_bounds__(256)
gemm_f16(const _Float16* __restrict__ A, const _Float16* __restrict__ Wt,
         const float* __restrict__ bias, float* __restrict__ outF,
         _Float16* __restrict__ outH, int M, int N, int K, int relu) {
  __shared__ __align__(16) _Float16 As[2][128 * AP];
  __shared__ __align__(16) _Float16 Bs[2][128 * AP];
  int tid = threadIdx.x;
  int wave = tid >> 5;
  int lane = tid & 31;
  int hl = lane >> 4, nl = lane & 15;
  long tile_m = (long)blockIdx.y * 128;
  long tile_n = (long)blockIdx.x * 128;
  v8f acc[8] = {};

  int lr = tid >> 1, lc = (tid & 1) * 16;  // 128 rows x 32 halfs, 16 halfs/thread
  const _Float16* gA = A + (tile_m + lr) * (long)K + lc;
  const _Float16* gB = Wt + (tile_n + lr) * (long)K + lc;
  int lofs = lr * AP + lc;

#if USE_ASYNC
  auto issue = [&](int k0, int buf) {
    async_b128(gA + k0, &As[buf][lofs]);
    async_b128(gA + k0 + 8, &As[buf][lofs + 8]);
    async_b128(gB + k0, &Bs[buf][lofs]);
    async_b128(gB + k0 + 8, &Bs[buf][lofs + 8]);
  };
  issue(0, 0);
  if (K > 32) issue(32, 1);
  for (int k0 = 0; k0 < K; k0 += 32) {
    int cur = (k0 >> 5) & 1;
    if (k0 + 32 < K) __builtin_amdgcn_s_wait_asynccnt(4);
    else             __builtin_amdgcn_s_wait_asynccnt(0);
    __syncthreads();
    v16h af = frag_a(&As[cur][wave * 16 * AP], AP);
#pragma unroll
    for (int j = 0; j < 8; ++j)
      acc[j] = wmma32(af, frag_b_t(&Bs[cur][j * 16 * AP], AP), acc[j]);
    __syncthreads();
    if (k0 + 64 < K) issue(k0 + 64, cur);
  }
#else
  v8h ra0, ra1, rb0, rb1;
  auto gload = [&](int k0) {
    const v8h* pa = (const v8h*)(gA + k0);
    ra0 = pa[0]; ra1 = pa[1];
    const v8h* pb = (const v8h*)(gB + k0);
    rb0 = pb[0]; rb1 = pb[1];
  };
  auto lstore = [&](int buf) {
    *(v8h*)(&As[buf][lofs]) = ra0; *(v8h*)(&As[buf][lofs + 8]) = ra1;
    *(v8h*)(&Bs[buf][lofs]) = rb0; *(v8h*)(&Bs[buf][lofs + 8]) = rb1;
  };
  gload(0);
  lstore(0);
  if (K > 32) gload(32);
  __syncthreads();
  for (int k0 = 0; k0 < K; k0 += 32) {
    int cur = (k0 >> 5) & 1;
    v16h af = frag_a(&As[cur][wave * 16 * AP], AP);
#pragma unroll
    for (int j = 0; j < 8; ++j)
      acc[j] = wmma32(af, frag_b_t(&Bs[cur][j * 16 * AP], AP), acc[j]);
    if (k0 + 32 < K) {
      lstore(cur ^ 1);
      if (k0 + 64 < K) gload(k0 + 64);
    }
    __syncthreads();
  }
#endif

#pragma unroll
  for (int j = 0; j < 8; ++j) {
    long col = tile_n + j * 16 + nl;
    float bv = bias ? bias[col] : 0.0f;
#pragma unroll
    for (int v = 0; v < 8; ++v) {
      long row = tile_m + wave * 16 + v + 8 * hl;
      float val = acc[j][v] + bv;
      if (relu) val = fmaxf(val, 0.0f);
      if (outF) outF[row * (long)N + col] = val;
      if (outH) outH[row * (long)N + col] = (_Float16)val;
    }
  }
}

// ---- two-pass flash attention: 4 waves/block, 64 q rows, shared K/V tiles --
// Q/K: [B*T, H*64] f16.  Vt: [B,H,64,S] f16.  mode 0: mask[b,q,s]; 1: mask[b,s]
#define KP 72   // K tile pitch (64-wide rows)
__global__ void __launch_bounds__(128)
attn_kernel(const _Float16* __restrict__ Qh, const _Float16* __restrict__ Kh,
            const _Float16* __restrict__ Vt, const int* __restrict__ mask,
            float* __restrict__ probs, _Float16* __restrict__ ctx,
            int Tq, int Sk, int mode) {
  __shared__ __align__(16) _Float16 Qs[4][16 * KP];
  __shared__ __align__(16) _Float16 Ks[32 * KP];   // [s][d] (N-major for QK^T)
  __shared__ __align__(16) _Float16 Vts[64 * AP];  // [d][s] (N-major for P.V)
  __shared__ __align__(16) _Float16 Ps[4][16 * AP];
  __shared__ float Ss[4][16 * 33];
  __shared__ float rowm_s[4][16], rowl_s[4][16];

  int tid = threadIdx.x;
  int wave = tid >> 5;
  int lane = tid & 31;
  int hl = lane >> 4, nl = lane & 15;
  int bh = blockIdx.y;
  int b = bh / NHEAD, h = bh % NHEAD;
  int q0 = blockIdx.x * 64 + wave * 16;

  {  // per-wave Q tile 16x64
    int r = lane >> 1, c = (lane & 1) * 32;
    const v8h* g =
        (const v8h*)(Qh + ((long)(b * Tq + q0 + r)) * D_MODEL + h * DHEAD + c);
    v8h t0 = g[0], t1 = g[1], t2 = g[2], t3 = g[3];
    _Float16* qd = &Qs[wave][r * KP + c];
    *(v8h*)(qd) = t0; *(v8h*)(qd + 8) = t1;
    *(v8h*)(qd + 16) = t2; *(v8h*)(qd + 24) = t3;
  }
  __syncthreads();
  v16h qf0 = frag_a(&Qs[wave][0], KP);   // dk 0..31
  v16h qf1 = frag_a(&Qs[wave][32], KP);  // dk 32..63

  float m_r = -1e30f, l_r = 0.0f;
  v8f o[4] = {};

  for (int pass = 0; pass < 2; ++pass) {
    if (pass == 1 && lane < 16) { rowm_s[wave][lane] = m_r; rowl_s[wave][lane] = l_r; }
    __syncthreads();
    for (int s0 = 0; s0 < Sk; s0 += 32) {
      {  // cooperative: K chunk 32x64; V^T chunk 64x32 (pass 2 only)
        int kr = tid >> 2, kc = (tid & 3) * 16;
        const v8h* gk = (const v8h*)(Kh + ((long)(b * Sk + s0 + kr)) * D_MODEL +
                                     h * DHEAD + kc);
        v8h k0v = gk[0], k1v = gk[1];
        *(v8h*)(Ks + kr * KP + kc) = k0v;
        *(v8h*)(Ks + kr * KP + kc + 8) = k1v;
        if (pass) {
          int vr = tid >> 1, vc = (tid & 1) * 16;
          const v8h* gv = (const v8h*)(Vt + ((long)bh * DHEAD + vr) * Sk + s0 + vc);
          v8h v0 = gv[0], v1 = gv[1];
          *(v8h*)(Vts + vr * AP + vc) = v0;
          *(v8h*)(Vts + vr * AP + vc + 8) = v1;
        }
      }
      __syncthreads();
#pragma unroll
      for (int t = 0; t < 2; ++t) {  // two 16x16 score tiles per 32-chunk
        v8f sc = {};
        sc = wmma32(qf0, frag_b_t(Ks + t * 16 * KP, KP), sc);
        sc = wmma32(qf1, frag_b_t(Ks + t * 16 * KP + 32, KP), sc);
#pragma unroll
        for (int v = 0; v < 8; ++v) {
          int r = v + 8 * hl;
          int s = s0 + t * 16 + nl;
          int q = q0 + r;
          float e = sc[v] * 0.125f;  // 1/sqrt(64)
          int mv = mode ? mask[(long)b * Sk + s]
                        : mask[((long)b * Tq + q) * Sk + s];
          if (mv == 0) e = -10000.0f;
          if (pass == 0) {
            Ss[wave][r * 33 + t * 16 + nl] = e;
          } else {
            float lr = rowl_s[wave][r];
            float p = (lr > 0.0f) ? __expf(e - rowm_s[wave][r]) / lr : 0.0f;
            Ps[wave][r * AP + t * 16 + nl] = (_Float16)p;
            if (probs) probs[((long)bh * Tq + q) * Sk + s] = p;
          }
        }
      }
      __syncthreads();
      if (pass == 0) {  // online row max / sum (per wave, 16 rows)
        if (lane < 16) {
          float cmax = -1e30f;
#pragma unroll
          for (int i = 0; i < 32; ++i)
            cmax = fmaxf(cmax, Ss[wave][lane * 33 + i]);
          float nm = fmaxf(m_r, cmax);
          float acc = 0.0f;
#pragma unroll
          for (int i = 0; i < 32; ++i)
            acc += __expf(Ss[wave][lane * 33 + i] - nm);
          l_r = l_r * __expf(m_r - nm) + acc;
          m_r = nm;
        }
      } else {  // O += P @ V
        v16h pf = frag_a(&Ps[wave][0], AP);
#pragma unroll
        for (int j = 0; j < 4; ++j)
          o[j] = wmma32(pf, frag_b_t(Vts + j * 16 * AP, AP), o[j]);
      }
      __syncthreads();
    }
  }
#pragma unroll
  for (int j = 0; j < 4; ++j) {
#pragma unroll
    for (int v = 0; v < 8; ++v) {
      int r = v + 8 * hl;
      ctx[((long)(b * Tq + q0 + r)) * D_MODEL + h * DHEAD + j * 16 + nl] =
          (_Float16)o[j][v];
    }
  }
}

// ---- fused residual add + LayerNorm (one block per row of 1024) ------------
__global__ void __launch_bounds__(256)
add_ln(const float* __restrict__ x, const float* __restrict__ r,
       const float* __restrict__ g, const float* __restrict__ bta,
       float* __restrict__ outF, _Float16* __restrict__ outH) {
  __shared__ float red[256];
  int row = blockIdx.x, tid = threadIdx.x;
  float vals[4];
  float s = 0.0f;
#pragma unroll
  for (int i = 0; i < 4; ++i) {
    int c = tid + i * 256;
    vals[i] = x[(long)row * 1024 + c] + r[(long)row * 1024 + c];
    s += vals[i];
  }
  red[tid] = s;
  __syncthreads();
  for (int w = 128; w > 0; w >>= 1) {
    if (tid < w) red[tid] += red[tid + w];
    __syncthreads();
  }
  float mean = red[0] / 1024.0f;
  __syncthreads();
  float s2 = 0.0f;
#pragma unroll
  for (int i = 0; i < 4; ++i) { float d = vals[i] - mean; s2 += d * d; }
  red[tid] = s2;
  __syncthreads();
  for (int w = 128; w > 0; w >>= 1) {
    if (tid < w) red[tid] += red[tid + w];
    __syncthreads();
  }
  float inv = rsqrtf(red[0] / 1024.0f + EPS_);
#pragma unroll
  for (int i = 0; i < 4; ++i) {
    int c = tid + i * 256;
    float y = (vals[i] - mean) * inv * g[c] + bta[c];
    if (outF) outF[(long)row * 1024 + c] = y;
    if (outH) outH[(long)row * 1024 + c] = (_Float16)y;
  }
}

extern "C" void kernel_launch(void* const* d_in, const int* in_sizes, int n_in,
                              void* d_out, int out_size, void* d_ws, size_t ws_size,
                              hipStream_t stream) {
  (void)in_sizes; (void)n_in; (void)out_size; (void)ws_size;
  const float* trg      = (const float*)d_in[0];
  const float* enc_src  = (const float*)d_in[1];
  const int*   src_mask = (const int*)d_in[2];
  const int*   trg_mask = (const int*)d_in[3];
  const float* sa_wq = (const float*)d_in[4];  const float* sa_bq = (const float*)d_in[5];
  const float* sa_wk = (const float*)d_in[6];  const float* sa_bk = (const float*)d_in[7];
  const float* sa_wv = (const float*)d_in[8];  const float* sa_bv = (const float*)d_in[9];
  const float* sa_wo = (const float*)d_in[10]; const float* sa_bo = (const float*)d_in[11];
  const float* ca_wq = (const float*)d_in[12]; const float* ca_bq = (const float*)d_in[13];
  const float* ca_wk = (const float*)d_in[14]; const float* ca_bk = (const float*)d_in[15];
  const float* ca_wv = (const float*)d_in[16]; const float* ca_bv = (const float*)d_in[17];
  const float* ca_wo = (const float*)d_in[18]; const float* ca_bo = (const float*)d_in[19];
  const float* ff_w1 = (const float*)d_in[20]; const float* ff_b1 = (const float*)d_in[21];
  const float* ff_w2 = (const float*)d_in[22]; const float* ff_b2 = (const float*)d_in[23];
  const float* ln1_g = (const float*)d_in[24]; const float* ln1_b = (const float*)d_in[25];
  const float* ln2_g = (const float*)d_in[26]; const float* ln2_b = (const float*)d_in[27];
  const float* ln3_g = (const float*)d_in[28]; const float* ln3_b = (const float*)d_in[29];

  const long M = MROWS;  // 4096 token rows
  char* base = (char*)d_ws;
  size_t off = 0;
  auto alloc = [&](size_t bytes) -> void* {
    void* p = base + off;
    off += (bytes + 255) & ~(size_t)255;
    return p;
  };

  _Float16* h_x    = (_Float16*)alloc((size_t)M * 1024 * 2);  // trg_h / ln1_h / ln2_h
  _Float16* h_enc  = (_Float16*)alloc((size_t)M * 1024 * 2);
  _Float16* h_q    = (_Float16*)alloc((size_t)M * 1024 * 2);
  _Float16* h_k    = (_Float16*)alloc((size_t)M * 1024 * 2);
  _Float16* h_v    = (_Float16*)alloc((size_t)M * 1024 * 2);
  _Float16* h_vt   = (_Float16*)alloc((size_t)M * 1024 * 2);  // [B,H,64,S]
  _Float16* h_ctx  = (_Float16*)alloc((size_t)M * 1024 * 2);
  _Float16* w_saq  = (_Float16*)alloc((size_t)1024 * 1024 * 2);   // all W^T [N,K]
  _Float16* w_sak  = (_Float16*)alloc((size_t)1024 * 1024 * 2);
  _Float16* w_sav  = (_Float16*)alloc((size_t)1024 * 1024 * 2);
  _Float16* w_sao  = (_Float16*)alloc((size_t)1024 * 1024 * 2);
  _Float16* w_caq  = (_Float16*)alloc((size_t)1024 * 1024 * 2);
  _Float16* w_cak  = (_Float16*)alloc((size_t)1024 * 1024 * 2);
  _Float16* w_cav  = (_Float16*)alloc((size_t)1024 * 1024 * 2);
  _Float16* w_cao  = (_Float16*)alloc((size_t)1024 * 1024 * 2);
  _Float16* w_ff1  = (_Float16*)alloc((size_t)1024 * DFF_N * 2);
  _Float16* w_ff2  = (_Float16*)alloc((size_t)DFF_N * 1024 * 2);
  _Float16* h_ff1  = (_Float16*)alloc((size_t)M * DFF_N * 2);
  float*    f_gemm = (float*)alloc((size_t)M * 1024 * 4);
  float*    f_t1   = (float*)alloc((size_t)M * 1024 * 4);
  float*    f_t2   = (float*)alloc((size_t)M * 1024 * 4);

  auto conv = [&](const float* s, _Float16* d, long n) {
    f32_to_f16_k<<<dim3((unsigned)((n + 255) / 256)), 256, 0, stream>>>(s, d, n);
  };
  auto convT = [&](const float* s, _Float16* d, int K, int N) {
    f32_to_f16_tr<<<dim3(N / 32, K / 32), 256, 0, stream>>>(s, d, K, N);
  };
  auto gemm = [&](const _Float16* A, const _Float16* Wt, const float* bias,
                  float* oF, _Float16* oH, int Mr, int N, int K, int relu) {
    gemm_f16<<<dim3(N / 128, Mr / 128), 256, 0, stream>>>(A, Wt, bias, oF, oH, Mr, N, K, relu);
  };
  auto vtr = [&](const _Float16* s, _Float16* d) {
    vhead_tr<<<dim3(SEQ / 32, D_MODEL / 32, BATCH), 256, 0, stream>>>(s, d);
  };

  conv(trg, h_x, M * 1024);
  conv(enc_src, h_enc, M * 1024);
  convT(sa_wq, w_saq, 1024, 1024); convT(sa_wk, w_sak, 1024, 1024);
  convT(sa_wv, w_sav, 1024, 1024); convT(sa_wo, w_sao, 1024, 1024);
  convT(ca_wq, w_caq, 1024, 1024); convT(ca_wk, w_cak, 1024, 1024);
  convT(ca_wv, w_cav, 1024, 1024); convT(ca_wo, w_cao, 1024, 1024);
  convT(ff_w1, w_ff1, 1024, DFF_N); convT(ff_w2, w_ff2, DFF_N, 1024);

  // ---- self-attention + residual LN ----
  gemm(h_x, w_saq, sa_bq, nullptr, h_q, (int)M, 1024, 1024, 0);
  gemm(h_x, w_sak, sa_bk, nullptr, h_k, (int)M, 1024, 1024, 0);
  gemm(h_x, w_sav, sa_bv, nullptr, h_v, (int)M, 1024, 1024, 0);
  vtr(h_v, h_vt);
  attn_kernel<<<dim3(SEQ / 64, BATCH * NHEAD), 128, 0, stream>>>(
      h_q, h_k, h_vt, trg_mask, nullptr, h_ctx, SEQ, SEQ, 0);
  gemm(h_ctx, w_sao, sa_bo, f_gemm, nullptr, (int)M, 1024, 1024, 0);
  add_ln<<<dim3((unsigned)M), 256, 0, stream>>>(trg, f_gemm, ln1_g, ln1_b, f_t1, h_x);

  // ---- cross-attention + residual LN (attention probs -> d_out tail) ----
  gemm(h_x, w_caq, ca_bq, nullptr, h_q, (int)M, 1024, 1024, 0);
  gemm(h_enc, w_cak, ca_bk, nullptr, h_k, (int)M, 1024, 1024, 0);
  gemm(h_enc, w_cav, ca_bv, nullptr, h_v, (int)M, 1024, 1024, 0);
  vtr(h_v, h_vt);
  float* attn_out = (float*)d_out + (long)M * 1024;
  attn_kernel<<<dim3(SEQ / 64, BATCH * NHEAD), 128, 0, stream>>>(
      h_q, h_k, h_vt, src_mask, attn_out, h_ctx, SEQ, SEQ, 1);
  gemm(h_ctx, w_cao, ca_bo, f_gemm, nullptr, (int)M, 1024, 1024, 0);
  add_ln<<<dim3((unsigned)M), 256, 0, stream>>>(f_t1, f_gemm, ln2_g, ln2_b, f_t2, h_x);

  // ---- FFN + residual LN -> d_out head ----
  gemm(h_x, w_ff1, ff_b1, nullptr, h_ff1, (int)M, DFF_N, 1024, 1);
  gemm(h_ff1, w_ff2, ff_b2, f_gemm, nullptr, (int)M, 1024, DFF_N, 0);
  add_ln<<<dim3((unsigned)M), 256, 0, stream>>>(f_t2, f_gemm, ln3_g, ln3_b,
                                                (float*)d_out, nullptr);
}
